// MambaModel_24223615550402
// MI455X (gfx1250) — matrix operand order repaired and checked by reading, not compiled, not run on hardware
//
#include <hip/hip_runtime.h>
#include <hip/hip_bf16.h>
#include <cstddef>

// ---------------------------------------------------------------------------
// Mamba classifier forward: B=8, L=512, ENC_IN=21, D_MODEL=512, D_INNER=1024,
// DT_RANK=32, N_STATE=16, D_CONV=4, E_LAYERS=4, NUM_CLASS=10
// ---------------------------------------------------------------------------

#define KB        8
#define KL        512
#define KENC      21
#define KDM       512
#define KDI       1024
#define KDTR      32
#define KNS       16
#define KDCONV    4
#define KLAYERS   4
#define KNCLS     10
#define KEPS      1e-5f

#define KBL       (KB * KL)          // 4096 rows of activations
#define KXDBL     (KDTR + 2 * KNS)   // 64

// GEMM macro-tiling: block tile 128x64, wave tile 32x32 (2x2 WMMA subtiles)
#define TM        128
#define TN        64
#define TK        32
#define ALDS      33                 // padded LDS row stride for A (bank spread)

typedef __attribute__((ext_vector_type(16))) _Float16     v16h;
typedef __attribute__((ext_vector_type(8)))  float        v8f;
typedef __attribute__((ext_vector_type(4)))  unsigned int u32x4;
typedef __attribute__((ext_vector_type(8)))  int          i32x8;
typedef __attribute__((ext_vector_type(4)))  int          i32x4;

#if defined(__AMDGCN__) && __has_builtin(__builtin_amdgcn_tensor_load_to_lds)
#define USE_TDM 1
#else
#define USE_TDM 0
#endif

// ---------------------------------------------------------------------------
// WMMA GEMM: C[M,N] = A[M,K] * B[K,N], fp32 in/out, f16 MACs, f32 accumulate.
// Grid: (N/TN, M/TM); block 256 threads = 8 waves, each wave owns a 32x32
// output tile (2x2 v_wmma_f32_16x16x32_f16 accumulators).
// A tile (128x32 f32) staged cooperatively with coalesced float4 loads;
// B tile (32x64 f32, row stride ldb) staged by the Tensor Data Mover
// (tensor_load_to_lds, TENSORcnt) issued by wave 0 each K-step.
// All dims here are multiples of TM/TN/TK, so no edge handling and EXEC is
// all-1s everywhere (WMMA requirement).
// Fragment layouts used here (wave32; contiguous K, not the ISA table order):
//   A (16x32 f16): lane&15 = M; element e -> K = e + 16*(lane>=16)
//   B (32x16 f16): lane&15 = N; element e -> K = e + 16*(lane>=16)
//   C (16x16 f32): vgpr r -> M = r + 8*(lane>=16); lane&15 = N
// ---------------------------------------------------------------------------
__global__ void wmma_gemm_f16(const float* __restrict__ A,
                              const float* __restrict__ Bw,
                              float* __restrict__ C,
                              int K, int lda, int ldb, int ldc) {
    __shared__ float As[TM * ALDS];      // 128 x 32 (+pad)
    __shared__ float Bs[TK * TN];        // 32 x 64

    const int tid  = threadIdx.x;
    const int lane = tid & 31;
    const int wid  = tid >> 5;           // wave 0..7
    const int mw   = wid >> 1;           // 0..3 : wave row in block tile
    const int nw   = wid & 1;            // 0..1 : wave col in block tile
    const int hi   = lane >> 4;          // half-wave select
    const int lr   = lane & 15;

    const int m0 = blockIdx.y * TM;
    const int n0 = blockIdx.x * TN;

    v8f acc[2][2] = {};

    for (int k0 = 0; k0 < K; k0 += TK) {
        __syncthreads();                 // previous compute done before overwrite

        // ---- stage A tile: 128x32 f32, 4 float4 per thread, coalesced ----
#pragma unroll
        for (int j = 0; j < (TM * TK) / (256 * 4); ++j) {
            const int f4 = tid + j * 256;
            const int r  = f4 >> 3;              // 8 float4 per 32-col row
            const int c4 = (f4 & 7) << 2;
            const float4 v =
                *(const float4*)(A + (size_t)(m0 + r) * lda + k0 + c4);
            float* dst = &As[r * ALDS + c4];
            dst[0] = v.x; dst[1] = v.y; dst[2] = v.z; dst[3] = v.w;
        }
        // prefetch next A tile while this K-step computes
        if (k0 + TK < K)
            __builtin_prefetch(A + (size_t)(m0 + (tid >> 3)) * lda + k0 + TK, 0, 0);

#if USE_TDM
        // ---- stage B tile via Tensor Data Mover (wave 0 issues the DMA) ----
        if (tid < 32) {
            const unsigned long long gaddr =
                (unsigned long long)(const void*)(Bw + (size_t)k0 * ldb + n0);
            const unsigned ldsb = (unsigned)(size_t)(void*)Bs; // addr[31:0] = LDS offset

            u32x4 g0;
            g0.x = 1u;                                   // count=1, user D#
            g0.y = ldsb;                                 // lds_addr (bits 63:32)
            g0.z = (unsigned)gaddr;                      // global_addr (bits 95:64)
            g0.w = (unsigned)((gaddr >> 32) & 0x1FFFFFFu)// global_addr (bits 120:96)
                 | (2u << 30);                           // type=2 ("image")

            i32x8 g1;
            g1[0] = (int)(2u << 16);                     // data_size = 4 bytes
            g1[1] = (int)(((unsigned)ldb & 0xFFFFu) << 16);        // tensor_dim0 lo16
            g1[2] = (int)((((unsigned)K & 0xFFFFu) << 16)          // tensor_dim1 lo16
                          | (((unsigned)ldb >> 16) & 0xFFFFu));    // tensor_dim0 hi16
            g1[3] = (int)((unsigned)TN << 16);           // tile_dim0 = 64
            g1[4] = (int)TK;                             // tile_dim1 = 32
            g1[5] = (int)ldb;                            // tensor_dim0_stride lo32
            g1[6] = 0;                                   // stride hi / dim1_stride
            g1[7] = 0;

            const i32x4 gz = {0, 0, 0, 0};
#if __clang_major__ >= 23
            const i32x8 gz8 = {0, 0, 0, 0, 0, 0, 0, 0};
            __builtin_amdgcn_tensor_load_to_lds(g0, g1, gz, gz, gz8, 0);
#else
            __builtin_amdgcn_tensor_load_to_lds(g0, g1, gz, gz, 0);
#endif
            __builtin_amdgcn_s_wait_tensorcnt(0);
        }
#else
        // ---- fallback: cooperative B tile load (also used by host pass) ----
#pragma unroll
        for (int j = 0; j < (TK * TN) / 256; ++j) {
            const int e = tid + j * 256;
            const int r = e >> 6;
            const int c = e & 63;
            Bs[r * TN + c] = Bw[(size_t)(k0 + r) * ldb + n0 + c];
        }
#endif
        __syncthreads();                 // tiles visible to all waves

        // ---- build fragments from LDS and issue 4 WMMAs ----
        v16h af[2];
#pragma unroll
        for (int i = 0; i < 2; ++i) {
            const int rowL = mw * 32 + i * 16 + lr;
#pragma unroll
            for (int e = 0; e < 16; ++e) {
                const int ka = e + (hi << 4);
                af[i][e] = (_Float16)As[rowL * ALDS + ka];
            }
        }
#pragma unroll
        for (int jn = 0; jn < 2; ++jn) {
            const int colL = nw * 32 + jn * 16 + lr;
            v16h bf;
#pragma unroll
            for (int e = 0; e < 16; ++e) {
                const int kb = e + (hi << 4);
                bf[e] = (_Float16)Bs[kb * TN + colL];
            }
#pragma unroll
            for (int i = 0; i < 2; ++i)
                acc[i][jn] = __builtin_amdgcn_wmma_f32_16x16x32_f16(
                    /*neg_a=*/false, af[i], /*neg_b=*/false, bf,
                    /*c_mod=*/(short)0, acc[i][jn],
                    /*reuse_a=*/false, /*reuse_b=*/false);
        }
    }

    // ---- store 2x2 accumulator tiles ----
#pragma unroll
    for (int i = 0; i < 2; ++i) {
#pragma unroll
        for (int jn = 0; jn < 2; ++jn) {
            float* crow = C + (size_t)(m0 + mw * 32 + i * 16 + (hi << 3)) * ldc
                            + n0 + nw * 32 + jn * 16 + lr;
#pragma unroll
            for (int r = 0; r < 8; ++r)
                crow[(size_t)r * ldc] = acc[i][jn][r];
        }
    }
}

// ---------------------------------------------------------------------------
// Token embedding: circular conv1d (ENC_IN -> D_MODEL, k=3, wrap pad) fused
// with sinusoidal positional embedding.
// ---------------------------------------------------------------------------
__global__ void token_embed_kernel(const float* __restrict__ xe,
                                   const float* __restrict__ tw,
                                   float* __restrict__ x) {
    const int idx = blockIdx.x * blockDim.x + threadIdx.x;
    if (idx >= KBL * KDM) return;
    const int d  = idx % KDM;
    const int bl = idx / KDM;
    const int l  = bl % KL;
    const int b  = bl / KL;

    float acc = 0.f;
#pragma unroll
    for (int j = 0; j < 3; ++j) {
        int lt = l - 1 + j;
        lt = (lt + KL) % KL;                               // circular pad
        const float* xp = xe + ((size_t)b * KL + lt) * KENC;
        const float* wp = tw + (size_t)d * KENC * 3 + j;
        for (int c = 0; c < KENC; ++c)
            acc += xp[c] * wp[c * 3];
    }
    // interleaved sin/cos positional embedding
    const int   i2  = d & ~1;
    const float dv  = __expf(-(float)i2 * (9.210340372f / (float)KDM)); // ln(1e4)
    const float ang = (float)l * dv;
    acc += (d & 1) ? __cosf(ang) : __sinf(ang);
    x[idx] = acc;
}

// ---------------------------------------------------------------------------
// RMSNorm over last dim D (one block per row).
// ---------------------------------------------------------------------------
__global__ void rmsnorm_kernel(const float* __restrict__ x,
                               const float* __restrict__ w,
                               float* __restrict__ out, int D) {
    __shared__ float red[256];
    const int row = blockIdx.x;
    const float* xr = x + (size_t)row * D;
    float s = 0.f;
    for (int i = threadIdx.x; i < D; i += blockDim.x) {
        const float v = xr[i];
        s += v * v;
    }
    red[threadIdx.x] = s;
    __syncthreads();
    for (int off = 128; off > 0; off >>= 1) {
        if (threadIdx.x < off) red[threadIdx.x] += red[threadIdx.x + off];
        __syncthreads();
    }
    const float scale = rsqrtf(red[0] / (float)D + KEPS);
    for (int i = threadIdx.x; i < D; i += blockDim.x)
        out[(size_t)row * D + i] = xr[i] * scale * w[i];
}

// ---------------------------------------------------------------------------
// Depthwise causal conv (k=4, pad left 3) + bias + SiLU on xs = xr[:, :, :DI].
// ---------------------------------------------------------------------------
__global__ void conv_silu_kernel(const float* __restrict__ xr,
                                 const float* __restrict__ w,
                                 const float* __restrict__ bias,
                                 float* __restrict__ xc) {
    const int idx = blockIdx.x * blockDim.x + threadIdx.x;
    if (idx >= KBL * KDI) return;
    const int d  = idx % KDI;
    const int bl = idx / KDI;
    const int l  = bl % KL;
    const int b  = bl / KL;

    float acc = bias[d];
#pragma unroll
    for (int j = 0; j < KDCONV; ++j) {
        const int lt = l + j - (KDCONV - 1);
        if (lt >= 0)
            acc += w[d * KDCONV + j] * xr[((size_t)b * KL + lt) * (2 * KDI) + d];
    }
    xc[idx] = acc / (1.f + __expf(-acc));                  // SiLU
}

// ---------------------------------------------------------------------------
// delta = softplus(delta_raw + dt_b[d])
// ---------------------------------------------------------------------------
__global__ void bias_softplus_kernel(float* __restrict__ delta,
                                     const float* __restrict__ dtb, int total) {
    const int idx = blockIdx.x * blockDim.x + threadIdx.x;
    if (idx >= total) return;
    const float v = delta[idx] + dtb[idx % KDI];
    delta[idx] = (v > 20.f) ? v : log1pf(__expf(v));
}

// ---------------------------------------------------------------------------
// Selective scan fused with output gate: one lane per (b, d) channel,
// 16-wide state in registers, sequential loop over L. B/C vectors are
// wave-uniform loads (all lanes in a wave share b,t) -> L2 broadcast.
// y[b,t,d] = (sum_n x_n * C_n + u * D[d]) * silu(res[b,t,d])
// ---------------------------------------------------------------------------
__global__ void scan_gate_kernel(const float* __restrict__ xc,
                                 const float* __restrict__ delta,
                                 const float* __restrict__ xdbl,
                                 const float* __restrict__ Alog,
                                 const float* __restrict__ Dp,
                                 const float* __restrict__ xr,
                                 float* __restrict__ y) {
    const int t = blockIdx.x * blockDim.x + threadIdx.x;
    if (t >= KB * KDI) return;
    const int b = t / KDI;
    const int d = t - b * KDI;

    float a[KNS], st[KNS];
#pragma unroll
    for (int n = 0; n < KNS; ++n) {
        a[n]  = -__expf(Alog[d * KNS + n]);
        st[n] = 0.f;
    }
    const float Dv = Dp[d];

    for (int l = 0; l < KL; ++l) {
        const size_t base = (size_t)b * KL + l;
        const float dt  = delta[base * KDI + d];
        const float u   = xc[base * KDI + d];
        const float* bc = xdbl + base * KXDBL;
        float yt = 0.f;
#pragma unroll
        for (int n = 0; n < KNS; ++n) {
            const float dA = __expf(dt * a[n]);
            st[n] = dA * st[n] + (dt * bc[KDTR + n]) * u;  // Bm at cols 32..47
            yt += st[n] * bc[KDTR + KNS + n];              // Cm at cols 48..63
        }
        const float res  = xr[base * (2 * KDI) + KDI + d];
        const float gate = res / (1.f + __expf(-res));     // SiLU(res)
        y[base * KDI + d] = (yt + u * Dv) * gate;
    }
}

// ---------------------------------------------------------------------------
// x += yo (residual)
// ---------------------------------------------------------------------------
__global__ void add_kernel(float* __restrict__ x, const float* __restrict__ yo,
                           int total) {
    const int idx = blockIdx.x * blockDim.x + threadIdx.x;
    if (idx < total) x[idx] += yo[idx];
}

// ---------------------------------------------------------------------------
// In-place exact GELU
// ---------------------------------------------------------------------------
__global__ void gelu_kernel(float* __restrict__ x, int total) {
    const int idx = blockIdx.x * blockDim.x + threadIdx.x;
    if (idx >= total) return;
    const float v = x[idx];
    x[idx] = 0.5f * v * (1.f + erff(v * 0.70710678f));
}

// ---------------------------------------------------------------------------
// Head: out[b,c] = sum_i feat[b,i] * proj_w[i,c] + proj_b[c]
// One block per (b,c), 256-thread reduction over i = 0..L*D_MODEL-1.
// ---------------------------------------------------------------------------
__global__ void head_kernel(const float* __restrict__ feat,
                            const float* __restrict__ pw,
                            const float* __restrict__ pb,
                            float* __restrict__ out) {
    __shared__ float red[256];
    const int b = blockIdx.x / KNCLS;
    const int c = blockIdx.x % KNCLS;
    const float* f = feat + (size_t)b * (KL * KDM);
    float s = 0.f;
    for (int i = threadIdx.x; i < KL * KDM; i += 256)
        s += f[i] * pw[(size_t)i * KNCLS + c];
    red[threadIdx.x] = s;
    __syncthreads();
    for (int off = 128; off > 0; off >>= 1) {
        if (threadIdx.x < off) red[threadIdx.x] += red[threadIdx.x + off];
        __syncthreads();
    }
    if (threadIdx.x == 0) out[b * KNCLS + c] = red[0] + pb[c];
}

// ---------------------------------------------------------------------------
// Host-side orchestration (graph-capture safe: only kernel launches on stream)
// ---------------------------------------------------------------------------
extern "C" void kernel_launch(void* const* d_in, const int* in_sizes, int n_in,
                              void* d_out, int out_size, void* d_ws, size_t ws_size,
                              hipStream_t stream) {
    const float* x_enc        = (const float*)d_in[0];
    const float* tok_w        = (const float*)d_in[1];
    const float* in_proj_w    = (const float*)d_in[2];
    const float* conv_w       = (const float*)d_in[3];
    const float* conv_b       = (const float*)d_in[4];
    const float* x_proj_w     = (const float*)d_in[5];
    const float* dt_w         = (const float*)d_in[6];
    const float* dt_b         = (const float*)d_in[7];
    const float* A_log        = (const float*)d_in[8];
    const float* D_p          = (const float*)d_in[9];
    const float* out_proj_w   = (const float*)d_in[10];
    const float* norm_w       = (const float*)d_in[11];
    const float* final_norm_w = (const float*)d_in[12];
    const float* proj_w       = (const float*)d_in[13];
    const float* proj_b       = (const float*)d_in[14];

    // Workspace layout (floats)
    float* ws   = (float*)d_ws;
    float* X    = ws;                                 // [4096, 512]
    float* XN   = X    + (size_t)KBL * KDM;           // [4096, 512]
    float* XR   = XN   + (size_t)KBL * KDM;           // [4096, 2048]
    float* XC   = XR   + (size_t)KBL * 2 * KDI;       // [4096, 1024]
    float* XDBL = XC   + (size_t)KBL * KDI;           // [4096, 64]
    float* DLT  = XDBL + (size_t)KBL * KXDBL;         // [4096, 1024]
    float* Y    = DLT  + (size_t)KBL * KDI;           // [4096, 1024]
    float* YO   = Y    + (size_t)KBL * KDI;           // [4096, 512]

    const int TPB = 256;
    const int nBLDM = KBL * KDM;   // 2,097,152
    const int nBLDI = KBL * KDI;   // 4,194,304

    // 1) token embedding + positional encoding
    token_embed_kernel<<<(nBLDM + TPB - 1) / TPB, TPB, 0, stream>>>(x_enc, tok_w, X);

    // 2) Mamba layers
    for (int i = 0; i < KLAYERS; ++i) {
        rmsnorm_kernel<<<KBL, TPB, 0, stream>>>(X, norm_w + (size_t)i * KDM, XN, KDM);

        // in_proj: [4096,512] x [512,2048] -> XR
        wmma_gemm_f16<<<dim3((2 * KDI) / TN, KBL / TM), TPB, 0, stream>>>(
            XN, in_proj_w + (size_t)i * KDM * 2 * KDI, XR,
            KDM, KDM, 2 * KDI, 2 * KDI);

        conv_silu_kernel<<<(nBLDI + TPB - 1) / TPB, TPB, 0, stream>>>(
            XR, conv_w + (size_t)i * KDI * KDCONV, conv_b + (size_t)i * KDI, XC);

        // x_proj: [4096,1024] x [1024,64] -> XDBL
        wmma_gemm_f16<<<dim3(KXDBL / TN, KBL / TM), TPB, 0, stream>>>(
            XC, x_proj_w + (size_t)i * KDI * KXDBL, XDBL,
            KDI, KDI, KXDBL, KXDBL);

        // dt proj: XDBL[:, :32] x [32,1024] -> DLT
        wmma_gemm_f16<<<dim3(KDI / TN, KBL / TM), TPB, 0, stream>>>(
            XDBL, dt_w + (size_t)i * KDTR * KDI, DLT,
            KDTR, KXDBL, KDI, KDI);

        bias_softplus_kernel<<<(nBLDI + TPB - 1) / TPB, TPB, 0, stream>>>(
            DLT, dt_b + (size_t)i * KDI, nBLDI);

        // fused selective scan + D skip + SiLU gate  (8192 channels)
        scan_gate_kernel<<<(KB * KDI) / TPB, TPB, 0, stream>>>(
            XC, DLT, XDBL, A_log + (size_t)i * KDI * KNS, D_p + (size_t)i * KDI,
            XR, Y);

        // out_proj: [4096,1024] x [1024,512] -> YO
        wmma_gemm_f16<<<dim3(KDM / TN, KBL / TM), TPB, 0, stream>>>(
            Y, out_proj_w + (size_t)i * KDI * KDM, YO,
            KDI, KDI, KDM, KDM);

        add_kernel<<<(nBLDM + TPB - 1) / TPB, TPB, 0, stream>>>(X, YO, nBLDM);
    }

    // 3) final norm + GELU + head
    rmsnorm_kernel<<<KBL, TPB, 0, stream>>>(X, final_norm_w, XN, KDM);
    gelu_kernel<<<(nBLDM + TPB - 1) / TPB, TPB, 0, stream>>>(XN, nBLDM);
    head_kernel<<<KB * KNCLS, TPB, 0, stream>>>(XN, proj_w, proj_b, (float*)d_out);
}
